// SharedMoE_29102698398030
// MI455X (gfx1250) — matrix operand order repaired and checked by reading, not compiled, not run on hardware
//
#include <hip/hip_runtime.h>

// Problem constants (match reference)
#define BB 4
#define SS 2048
#define HH 1024
#define EE 8
#define NN (BB * SS)        // 8192 tokens
#define RTILES (NN / 32)    // 256 routed 32-token tiles per expert (max)

// CDNA5 wave32 WMMA types
typedef __attribute__((ext_vector_type(16))) __bf16        bf16x16;
typedef __attribute__((ext_vector_type(8)))  float         f32x8;
typedef __attribute__((ext_vector_type(4)))  unsigned int  u32x4;
typedef __attribute__((ext_vector_type(4)))  float         f32x4;

union ABfrag { bf16x16 v; u32x4 q[2]; };

static __device__ __forceinline__ unsigned short f2bf(float f) {
  unsigned int u = __float_as_uint(f);
  u += 0x7fffu + ((u >> 16) & 1u);   // round-to-nearest-even
  return (unsigned short)(u >> 16);
}

static __device__ __forceinline__ f32x8 wmma_bf16(bf16x16 a, bf16x16 b, f32x8 c) {
  // 8 args: (neg_a, A, neg_b, B, c_mod, C, reuse_a, reuse_b)
  return __builtin_amdgcn_wmma_f32_16x16x32_bf16(false, a, false, b, (short)0, c,
                                                 false, false);
}

// ---------------- prep kernels (run once per launch, stream-ordered) --------

__global__ __launch_bounds__(32) void k_zero_counts(int* __restrict__ cnt) {
  if (threadIdx.x < EE) cnt[threadIdx.x] = 0;
}

__global__ __launch_bounds__(256) void k_convert_x(const float* __restrict__ x,
                                                   unsigned short* __restrict__ xbf) {
  size_t i = (size_t)blockIdx.x * 256 + threadIdx.x;   // N*H elements
  xbf[i] = f2bf(x[i]);
}

// W_sum^T[f][h] = bf16( sum_e W_shared[e][h][f] )
__global__ __launch_bounds__(256) void k_wsum_t(const float* __restrict__ Ws,
                                                unsigned short* __restrict__ wst) {
  size_t i = (size_t)blockIdx.x * 256 + threadIdx.x;   // H*H
  int h = (int)(i >> 10), f = (int)(i & (HH - 1));
  float s = 0.f;
#pragma unroll
  for (int e = 0; e < EE; ++e) s += Ws[(size_t)e * HH * HH + i];
  wst[(size_t)f * HH + h] = f2bf(s);
}

// W_routed^T[e][f][h] = bf16( W_routed[e][h][f] )
__global__ __launch_bounds__(256) void k_wroute_t(const float* __restrict__ Wr,
                                                  unsigned short* __restrict__ wrt) {
  size_t i = (size_t)blockIdx.x * 256 + threadIdx.x;   // E*H*H
  int e = (int)(i >> 20);
  size_t rem = i & ((size_t)HH * HH - 1);
  int h = (int)(rem >> 10), f = (int)(rem & (HH - 1));
  wrt[(size_t)e * HH * HH + (size_t)f * HH + h] = f2bf(Wr[i]);
}

__global__ __launch_bounds__(256) void k_bias_sum(const float* __restrict__ bs,
                                                  float* __restrict__ bsum) {
  int f = blockIdx.x * 256 + threadIdx.x;              // H
  float s = 0.f;
#pragma unroll
  for (int e = 0; e < EE; ++e) s += bs[(size_t)e * HH + f];
  bsum[f] = s;
}

// ---------------- gating: logits, softmax, top-2, expert dispatch -----------

__global__ __launch_bounds__(256) void k_gate(const float* __restrict__ x,
                                              const float* __restrict__ wg,   // [H][E]
                                              const float* __restrict__ bg,   // [E]
                                              float* __restrict__ logits,     // [N][E]
                                              int* __restrict__ cnt,          // [E]
                                              int* __restrict__ toks,         // [E][N]
                                              float* __restrict__ wts) {      // [E][N]
  const int n = blockIdx.x * 8 + threadIdx.y;          // one wave per token
  const int lane = threadIdx.x;
  const float* xr = x + (size_t)n * HH;

  float acc[EE] = {0.f, 0.f, 0.f, 0.f, 0.f, 0.f, 0.f, 0.f};
  for (int h = lane; h < HH; h += 32) {
    float xv = xr[h];
    f32x4 w0 = *(const f32x4*)(wg + (size_t)h * EE);
    f32x4 w1 = *(const f32x4*)(wg + (size_t)h * EE + 4);
    acc[0] += xv * w0[0]; acc[1] += xv * w0[1];
    acc[2] += xv * w0[2]; acc[3] += xv * w0[3];
    acc[4] += xv * w1[0]; acc[5] += xv * w1[1];
    acc[6] += xv * w1[2]; acc[7] += xv * w1[3];
  }
#pragma unroll
  for (int off = 16; off > 0; off >>= 1) {
#pragma unroll
    for (int e = 0; e < EE; ++e) acc[e] += __shfl_xor(acc[e], off, 32);
  }
  if (lane == 0) {
    float lg[EE], mx = -3.0e38f;
#pragma unroll
    for (int e = 0; e < EE; ++e) {
      lg[e] = acc[e] + bg[e];
      logits[(size_t)n * EE + e] = lg[e];
      mx = fmaxf(mx, lg[e]);
    }
    float ex[EE], den = 0.f;
#pragma unroll
    for (int e = 0; e < EE; ++e) { ex[e] = __expf(lg[e] - mx); den += ex[e]; }
    int i0 = 0;
#pragma unroll
    for (int e = 1; e < EE; ++e) if (lg[e] > lg[i0]) i0 = e;   // ties -> lowest idx
    int i1 = (i0 == 0) ? 1 : 0;
#pragma unroll
    for (int e = 0; e < EE; ++e) if (e != i0 && lg[e] > lg[i1]) i1 = e;
    float inv = 1.f / den;
    float p0 = ex[i0] * inv, p1 = ex[i1] * inv;
    int pos0 = atomicAdd(&cnt[i0], 1);
    toks[(size_t)i0 * NN + pos0] = n; wts[(size_t)i0 * NN + pos0] = p0;
    int pos1 = atomicAdd(&cnt[i1], 1);
    toks[(size_t)i1 * NN + pos1] = n; wts[(size_t)i1 * NN + pos1] = p1;
  }
}

// ---------------- shared GEMM: out = x_bf @ Wsum_bf + bias_sum --------------
// grid (N/32, H/512), block (32, 8); each wave -> 32 rows x 64 cols.
// 4 independent B buffers so all K-step loads issue before the first WMMA.

__global__ __launch_bounds__(256) void k_shared_gemm(
    const unsigned short* __restrict__ xbf,  // [N][H] bf16
    const unsigned short* __restrict__ wt,   // [H(f)][H(h)] bf16 (transposed)
    const float* __restrict__ bias,          // [H]
    float* __restrict__ out) {               // [N][H] f32
  const int lane = threadIdx.x, wave = threadIdx.y;
  const int n0 = blockIdx.x * 32;
  const int f0 = (blockIdx.y * 8 + wave) * 64;
  const int m = lane & 15, half = lane >> 4;

  const unsigned short* arow0 = xbf + (size_t)(n0 + m) * HH;
  const unsigned short* arow1 = xbf + (size_t)(n0 + 16 + m) * HH;
  const unsigned short* b0r = wt + (size_t)(f0 + m) * HH;       // col tile 0
  f32x8 cl0 = {}, cl1 = {}, cl2 = {}, cl3 = {};                 // rows n0..n0+15
  f32x8 ch0 = {}, ch1 = {}, ch2 = {}, ch3 = {};                 // rows n0+16..n0+31

  for (int k = 0; k < HH; k += 32) {
    __builtin_prefetch(arow0 + k + 256, 0, 3);
    __builtin_prefetch(b0r + k + 256, 0, 3);
    ABfrag al, ah, b0, b1, b2, b3;
    const unsigned short* bb = b0r + k + half * 8;
    al.q[0] = *(const u32x4*)(arow0 + k + half * 8);
    al.q[1] = *(const u32x4*)(arow0 + k + 16 + half * 8);
    ah.q[0] = *(const u32x4*)(arow1 + k + half * 8);
    ah.q[1] = *(const u32x4*)(arow1 + k + 16 + half * 8);
    b0.q[0] = *(const u32x4*)(bb);              b0.q[1] = *(const u32x4*)(bb + 16);
    b1.q[0] = *(const u32x4*)(bb + 16 * HH);    b1.q[1] = *(const u32x4*)(bb + 16 * HH + 16);
    b2.q[0] = *(const u32x4*)(bb + 32 * HH);    b2.q[1] = *(const u32x4*)(bb + 32 * HH + 16);
    b3.q[0] = *(const u32x4*)(bb + 48 * HH);    b3.q[1] = *(const u32x4*)(bb + 48 * HH + 16);
    cl0 = wmma_bf16(al.v, b0.v, cl0);
    ch0 = wmma_bf16(ah.v, b0.v, ch0);
    cl1 = wmma_bf16(al.v, b1.v, cl1);
    ch1 = wmma_bf16(ah.v, b1.v, ch1);
    cl2 = wmma_bf16(al.v, b2.v, cl2);
    ch2 = wmma_bf16(ah.v, b2.v, ch2);
    cl3 = wmma_bf16(al.v, b3.v, cl3);
    ch3 = wmma_bf16(ah.v, b3.v, ch3);
  }

  const float bi0 = bias[f0 + m],      bi1 = bias[f0 + 16 + m];
  const float bi2 = bias[f0 + 32 + m], bi3 = bias[f0 + 48 + m];
#pragma unroll
  for (int r = 0; r < 8; ++r) {
    float* o0 = out + (size_t)(n0 + r + 8 * half) * HH + f0 + m;
    o0[0]  = cl0[r] + bi0;
    o0[16] = cl1[r] + bi1;
    o0[32] = cl2[r] + bi2;
    o0[48] = cl3[r] + bi3;
    float* o1 = out + (size_t)(n0 + 16 + r + 8 * half) * HH + f0 + m;
    o1[0]  = ch0[r] + bi0;
    o1[16] = ch1[r] + bi1;
    o1[32] = ch2[r] + bi2;
    o1[48] = ch3[r] + bi3;
  }
}

// ---------------- routed GEMM: out += w * (x_bf @ W_e + b_e) ----------------
// grid (E*RTILES, H/512), block (32, 8); gathered 32-token tiles per expert.

__global__ __launch_bounds__(256) void k_routed_gemm(
    const unsigned short* __restrict__ xbf,  // [N][H] bf16
    const unsigned short* __restrict__ wrt,  // [E][H(f)][H(h)] bf16
    const float* __restrict__ br,            // [E][H]
    const int* __restrict__ cnt,
    const int* __restrict__ toks,            // [E][N]
    const float* __restrict__ wts,           // [E][N]
    float* __restrict__ out) {
  const int e = blockIdx.x / RTILES;
  const int t = blockIdx.x % RTILES;
  const int ce = cnt[e];
  const int base = t * 32;
  if (base >= ce) return;                    // tile beyond this expert's tokens

  const int lane = threadIdx.x, wave = threadIdx.y;
  const int m = lane & 15, half = lane >> 4;
  const int f0 = (blockIdx.y * 8 + wave) * 64;

  const int tl = base + m, th = base + 16 + m;
  const int tok0 = toks[(size_t)e * NN + (tl < ce ? tl : ce - 1)];
  const int tok1 = toks[(size_t)e * NN + (th < ce ? th : ce - 1)];
  const unsigned short* arow0 = xbf + (size_t)tok0 * HH;
  const unsigned short* arow1 = xbf + (size_t)tok1 * HH;
  const unsigned short* b0r =
      wrt + (size_t)e * HH * HH + (size_t)(f0 + m) * HH;
  f32x8 cl0 = {}, cl1 = {}, cl2 = {}, cl3 = {};
  f32x8 ch0 = {}, ch1 = {}, ch2 = {}, ch3 = {};

  for (int k = 0; k < HH; k += 32) {
    __builtin_prefetch(arow0 + k + 256, 0, 3);
    __builtin_prefetch(b0r + k + 256, 0, 3);
    ABfrag al, ah, b0, b1, b2, b3;
    const unsigned short* bb = b0r + k + half * 8;
    al.q[0] = *(const u32x4*)(arow0 + k + half * 8);
    al.q[1] = *(const u32x4*)(arow0 + k + 16 + half * 8);
    ah.q[0] = *(const u32x4*)(arow1 + k + half * 8);
    ah.q[1] = *(const u32x4*)(arow1 + k + 16 + half * 8);
    b0.q[0] = *(const u32x4*)(bb);              b0.q[1] = *(const u32x4*)(bb + 16);
    b1.q[0] = *(const u32x4*)(bb + 16 * HH);    b1.q[1] = *(const u32x4*)(bb + 16 * HH + 16);
    b2.q[0] = *(const u32x4*)(bb + 32 * HH);    b2.q[1] = *(const u32x4*)(bb + 32 * HH + 16);
    b3.q[0] = *(const u32x4*)(bb + 48 * HH);    b3.q[1] = *(const u32x4*)(bb + 48 * HH + 16);
    cl0 = wmma_bf16(al.v, b0.v, cl0);
    ch0 = wmma_bf16(ah.v, b0.v, ch0);
    cl1 = wmma_bf16(al.v, b1.v, cl1);
    ch1 = wmma_bf16(ah.v, b1.v, ch1);
    cl2 = wmma_bf16(al.v, b2.v, cl2);
    ch2 = wmma_bf16(ah.v, b2.v, ch2);
    cl3 = wmma_bf16(al.v, b3.v, cl3);
    ch3 = wmma_bf16(ah.v, b3.v, ch3);
  }

  const float bi0 = br[(size_t)e * HH + f0 + m];
  const float bi1 = br[(size_t)e * HH + f0 + 16 + m];
  const float bi2 = br[(size_t)e * HH + f0 + 32 + m];
  const float bi3 = br[(size_t)e * HH + f0 + 48 + m];
#pragma unroll
  for (int r = 0; r < 8; ++r) {
    const int tI0 = base + r + 8 * half;
    if (tI0 < ce) {
      const int row = toks[(size_t)e * NN + tI0];
      const float w = wts[(size_t)e * NN + tI0];
      float* orow = out + (size_t)row * HH + f0 + m;
      atomicAdd(orow + 0,  w * (cl0[r] + bi0));
      atomicAdd(orow + 16, w * (cl1[r] + bi1));
      atomicAdd(orow + 32, w * (cl2[r] + bi2));
      atomicAdd(orow + 48, w * (cl3[r] + bi3));
    }
    const int tI1 = base + 16 + r + 8 * half;
    if (tI1 < ce) {
      const int row = toks[(size_t)e * NN + tI1];
      const float w = wts[(size_t)e * NN + tI1];
      float* orow = out + (size_t)row * HH + f0 + m;
      atomicAdd(orow + 0,  w * (ch0[r] + bi0));
      atomicAdd(orow + 16, w * (ch1[r] + bi1));
      atomicAdd(orow + 32, w * (ch2[r] + bi2));
      atomicAdd(orow + 48, w * (ch3[r] + bi3));
    }
  }
}

// ---------------- host-side orchestration -----------------------------------

extern "C" void kernel_launch(void* const* d_in, const int* in_sizes, int n_in,
                              void* d_out, int out_size, void* d_ws, size_t ws_size,
                              hipStream_t stream) {
  const float* x  = (const float*)d_in[0];  // [B,S,H]
  const float* Ws = (const float*)d_in[1];  // [E,H,H]
  const float* bs = (const float*)d_in[2];  // [E,H]
  const float* Wg = (const float*)d_in[3];  // [H,E]
  const float* bg = (const float*)d_in[4];  // [E]
  const float* Wr = (const float*)d_in[5];  // [E,H,H]
  const float* br = (const float*)d_in[6];  // [E,H]

  float* out    = (float*)d_out;                       // [B,S,H]
  float* logits = out + (size_t)NN * HH;               // [N,E] second output

  // Workspace layout (~35 MB total), all 16B+ aligned
  char* ws = (char*)d_ws;
  unsigned short* xbf  = (unsigned short*)(ws);                        // 16 MB
  unsigned short* wst  = (unsigned short*)(ws + (16u << 20));          //  2 MB
  unsigned short* wrt  = (unsigned short*)(ws + (18u << 20));          // 16 MB
  float*          bsum = (float*)(ws + (34u << 20));                   //  4 KB
  int*            cntp = (int*)(ws + (34u << 20) + 4096);              // 32 B
  int*            toks = (int*)(ws + (34u << 20) + 8192);              // 256 KB
  float*          wtsb = (float*)(ws + (34u << 20) + 8192 + (size_t)EE * NN * 4);

  k_zero_counts<<<1, 32, 0, stream>>>(cntp);
  k_convert_x<<<(NN * HH) / 256, 256, 0, stream>>>(x, xbf);
  k_wsum_t<<<(HH * HH) / 256, 256, 0, stream>>>(Ws, wst);
  k_wroute_t<<<(EE * HH * HH) / 256, 256, 0, stream>>>(Wr, wrt);
  k_bias_sum<<<HH / 256, 256, 0, stream>>>(bs, bsum);
  k_gate<<<NN / 8, dim3(32, 8), 0, stream>>>(x, Wg, bg, logits, cntp, toks, wtsb);
  k_shared_gemm<<<dim3(NN / 32, HH / 512), dim3(32, 8), 0, stream>>>(xbf, wst, bsum, out);
  k_routed_gemm<<<dim3(EE * RTILES, HH / 512), dim3(32, 8), 0, stream>>>(
      xbf, wrt, br, cntp, toks, wtsb, out);
}
